// CNV_Binary_67018669687423
// MI455X (gfx1250) — 8-bit matrix operand order repaired and checked by reading, compiled, not run on hardware
//
#include <hip/hip_runtime.h>
#include <stdint.h>

typedef int v8i __attribute__((ext_vector_type(8)));

#define EPSV 1e-5f

__device__ __forceinline__ float sgnf(float x){ return (x>0.f)?1.f:((x<0.f)?-1.f:0.f); }
__device__ __forceinline__ int8_t sgn8(float x){ return (int8_t)((x>0.f)?1:((x<0.f)?-1:0)); }

__global__ void k_pack_sign(int8_t* __restrict__ dst, const float* __restrict__ src,
                            int Rout, int Rin, int Kout, int Kin) {
  long i = (long)blockIdx.x * 256 + threadIdx.x;
  long tot = (long)Rout * Kout;
  if (i >= tot) return;
  int r = (int)(i / Kout);
  int k = (int)(i - (long)r * Kout);
  int8_t v = 0;
  if (r < Rin && k < Kin) v = sgn8(src[(long)r * Kin + k]);
  dst[i] = v;
}

__global__ void k_pack_trip_f(int8_t* __restrict__ dst, const float* __restrict__ src,
                              int Rows, int Gin, int Gout, int Kin) {
  long i = (long)blockIdx.x * 256 + threadIdx.x;
  long tot = (long)Rows * Gout;
  if (i >= tot) return;
  int r = (int)(i / Gout);
  int g = (int)(i - (long)r * Gout);
  int8_t v = 0;
  if (g < Gin) {
    const float* s = src + (long)r * Kin + 3 * g;
    v = (int8_t)(sgn8(s[0]) * sgn8(s[1]) * sgn8(s[2]));
  }
  dst[i] = v;
}

__global__ void k_trip_i8(int8_t* __restrict__ dst, const int8_t* __restrict__ src,
                          int Rows, int Gin, int Gout, int Kin) {
  long i = (long)blockIdx.x * 256 + threadIdx.x;
  long tot = (long)Rows * Gout;
  if (i >= tot) return;
  int r = (int)(i / Gout);
  int g = (int)(i - (long)r * Gout);
  int8_t v = 0;
  if (g < Gin) {
    const int8_t* s = src + (long)r * Kin + 3 * g;
    v = (int8_t)(s[0] * s[1] * s[2]);
  }
  dst[i] = v;
}

__global__ void k_conv0(int8_t* __restrict__ a0, const float* __restrict__ x,
                        const float* __restrict__ w, const float* __restrict__ g,
                        const float* __restrict__ b, const float* __restrict__ m,
                        const float* __restrict__ vv) {
  int i = blockIdx.x * 256 + threadIdx.x;
  if (i >= 128 * 64 * 32 * 32) return;
  int ow = i & 31, oh = (i >> 5) & 31, f = (i >> 10) & 63, n = i >> 16;
  float acc = 0.f;
  #pragma unroll
  for (int c = 0; c < 3; c++)
    #pragma unroll
    for (int kh = 0; kh < 3; kh++) {
      int hh = oh + kh - 1;
      if (hh < 0 || hh >= 32) continue;
      #pragma unroll
      for (int kw = 0; kw < 3; kw++) {
        int ww = ow + kw - 1;
        if (ww < 0 || ww >= 32) continue;
        float xv = x[((n * 3 + c) * 32 + hh) * 32 + ww];
        float wv = w[((f * 3 + c) * 3 + kh) * 3 + kw];
        acc += xv * sgnf(wv);
      }
    }
  float s = g[f] * rsqrtf(vv[f] + EPSV);
  float t = m[f] - b[f] / s;
  a0[i] = (acc > t) ? (int8_t)1 : (int8_t)-1;
}

__global__ void k_x3(int8_t* __restrict__ dst, const int8_t* __restrict__ a,
                     int NC, int H, int W) {
  int i = blockIdx.x * 256 + threadIdx.x;
  if (i >= NC * H * W) return;
  int w = i % W;
  const int8_t* row = a + (i - w);
  int l = (w > 0)     ? (int)row[w - 1] : 0;
  int c = (int)row[w];
  int r = (w < W - 1) ? (int)row[w + 1] : 0;
  dst[i] = (int8_t)(l * c * r);
}

template <int KH, int KW>
__global__ __launch_bounds__(256)
void k_bgemm(int* __restrict__ Y, const int8_t* __restrict__ Xp,
             const int8_t* __restrict__ Bm, int NB, int C, int H, int W,
             int PH, int PW, int Kreal, int Kpad, int Npad) {
  __shared__ __align__(16) int8_t smA[64 * 64];
  __shared__ __align__(16) int8_t smB[64 * 64];
  const int t    = threadIdx.x;
  const int lane = t & 31;
  const int wv   = t >> 5;
  const int wm   = wv & 3;
  const int wn   = wv >> 2;
  const int m0   = blockIdx.x * 64;
  const int n0   = blockIdx.y * 64;

  const int mm = t >> 2;
  const int kc = (t & 3) * 16;
  const int HW = H * W;
  const int mrow  = m0 + mm;
  const int n_img = mrow / HW;
  const int rem   = mrow - n_img * HW;
  const int oh    = rem / W;
  const int ow    = rem - oh * W;
  constexpr int khkw = KH * KW;
  const int8_t* abase = Xp + (size_t)n_img * C * HW;

  const int half = lane >> 4;
  const int l15  = lane & 15;

  v8i acc0 = {0, 0, 0, 0, 0, 0, 0, 0};
  v8i acc1 = {0, 0, 0, 0, 0, 0, 0, 0};

  for (int k0 = 0; k0 < Kpad; k0 += 64) {
    if constexpr (KH == 1 && KW == 1) {
      *(int4*)(smA + mm * 64 + kc) = *(const int4*)(abase + k0 + kc);
    } else {
      __align__(16) int8_t tmp[16];
      #pragma unroll
      for (int j = 0; j < 16; j++) {
        int k  = k0 + kc + j;
        int kk = (k < Kreal) ? k : 0;
        int c  = kk / khkw;
        int r  = kk - c * khkw;
        int kh = r / KW;
        int kw = r - kh * KW;
        int hh = oh + kh - PH;
        int ww = ow + kw - PW;
        bool inb = (k < Kreal) & (hh >= 0) & (hh < H) & (ww >= 0) & (ww < W);
        int hc = min(max(hh, 0), H - 1);
        int wc = min(max(ww, 0), W - 1);
        int8_t raw = abase[(c * H + hc) * W + wc];
        tmp[j] = inb ? raw : (int8_t)0;
      }
      *(int4*)(smA + mm * 64 + kc) = *(const int4*)tmp;
    }

    const int8_t* bsrc = Bm + (size_t)(n0 + mm) * Kpad + k0 + kc;
    *(int4*)(smB + mm * 64 + kc) = *(const int4*)bsrc;
    if (k0 + 64 < Kpad) __builtin_prefetch(bsrc + 64, 0, 1);
    __syncthreads();

    v8i af, bf0, bf1;
    #pragma unroll
    for (int v = 0; v < 8; v++) {
      int ka = ((v & 1) << 2) + (((v >> 1) & 1) << 4) + ((v >> 2) << 5) + (half << 3);
      af[v] = *(const int*)(smA + (wm * 16 + l15) * 64 + ka);
      int kb = ((v & 1) << 2) + (((v >> 1) & 1) << 4) + ((v >> 2) << 5) + (half << 3);
      bf0[v] = *(const int*)(smB + (wn * 32 + l15) * 64 + kb);
      bf1[v] = *(const int*)(smB + (wn * 32 + 16 + l15) * 64 + kb);
    }
    acc0 = __builtin_amdgcn_wmma_i32_16x16x64_iu8(true, af, true, bf0, acc0, false, false);
    acc1 = __builtin_amdgcn_wmma_i32_16x16x64_iu8(true, af, true, bf1, acc1, false, false);
    __syncthreads();
  }

  #pragma unroll
  for (int r = 0; r < 8; r++) {
    int grow = m0 + wm * 16 + half * 8 + r;
    int gcol = n0 + wn * 32 + l15;
    Y[(size_t)grow * Npad + gcol]      = acc0[r];
    Y[(size_t)grow * Npad + gcol + 16] = acc1[r];
  }
}

__global__ void k_thresh_conv(int8_t* __restrict__ dst, const int* __restrict__ yf,
                              const int* __restrict__ yt, int NB, int F, int Fpad,
                              int H, int W, int pool, float scale,
                              const float* g, const float* b, const float* m,
                              const float* vv) {
  int Hp = H / pool, Wp = W / pool;
  long i = (long)blockIdx.x * 256 + threadIdx.x;
  long tot = (long)NB * F * Hp * Wp;
  if (i >= tot) return;
  int wp = (int)(i % Wp); long r = i / Wp;
  int hp = (int)(r % Hp); r /= Hp;
  int f = (int)(r % F);
  int n = (int)(r / F);
  float best = -3.4e38f;
  for (int dy = 0; dy < pool; dy++)
    for (int dx = 0; dx < pool; dx++) {
      int h = hp * pool + dy, w = wp * pool + dx;
      long mrow = ((long)n * H + h) * W + w;
      float v = (float)yf[mrow * Fpad + f];
      if (yt) v -= (float)yt[mrow * Fpad + f];
      v *= scale;
      if (v > best) best = v;
    }
  float s = g[f] * rsqrtf(vv[f] + EPSV);
  float t = m[f] - b[f] / s;
  dst[i] = (best > t) ? (int8_t)1 : (int8_t)-1;
}

__global__ void k_thresh_fc(int8_t* __restrict__ dst, const int* __restrict__ yf,
                            const int* __restrict__ yt, int N, int F, int Fpad,
                            float scale, const float* g, const float* b,
                            const float* m, const float* vv) {
  int i = blockIdx.x * 256 + threadIdx.x;
  if (i >= N * F) return;
  int f = i % F, n = i / F;
  float v = (float)yf[(size_t)n * Fpad + f];
  if (yt) v -= (float)yt[(size_t)n * Fpad + f];
  v *= scale;
  float s = g[f] * rsqrtf(vv[f] + EPSV);
  float t = m[f] - b[f] / s;
  dst[i] = (v > t) ? (int8_t)1 : (int8_t)-1;
}

__global__ void k_build_afc1(int8_t* __restrict__ dst, const int8_t* __restrict__ a5) {
  int i = blockIdx.x * 256 + threadIdx.x;
  if (i >= 128 * 4160) return;
  int k = i % 4160, n = i / 4160;
  int8_t v;
  if (k == 0 || k == 4097) v = -1;
  else if (k >= 4098)      v = 0;
  else                     v = a5[n * 4096 + (k - 1)];
  dst[i] = v;
}

__global__ __launch_bounds__(256)
void k_lsm(float* __restrict__ out, const int* __restrict__ y,
           const float* __restrict__ mu, const float* __restrict__ var,
           int C, int Cpad) {
  __shared__ float red[256];
  int n = blockIdx.x, t = threadIdx.x;
  float mx = -3.4e38f;
  for (int j = t; j < C; j += 256) {
    float z = ((float)y[(size_t)n * Cpad + j] - mu[j]) * rsqrtf(var[j] + EPSV);
    mx = fmaxf(mx, z);
  }
  red[t] = mx; __syncthreads();
  for (int s = 128; s > 0; s >>= 1) { if (t < s) red[t] = fmaxf(red[t], red[t + s]); __syncthreads(); }
  mx = red[0]; __syncthreads();
  float sm = 0.f;
  for (int j = t; j < C; j += 256) {
    float z = ((float)y[(size_t)n * Cpad + j] - mu[j]) * rsqrtf(var[j] + EPSV);
    sm += __expf(z - mx);
  }
  red[t] = sm; __syncthreads();
  for (int s = 128; s > 0; s >>= 1) { if (t < s) red[t] += red[t + s]; __syncthreads(); }
  float lse = mx + __logf(red[0]);
  for (int j = t; j < C; j += 256) {
    float z = ((float)y[(size_t)n * Cpad + j] - mu[j]) * rsqrtf(var[j] + EPSV);
    out[(size_t)n * C + j] = z - lse;
  }
}

extern "C" void kernel_launch(void* const* d_in, const int* in_sizes, int n_in,
                              void* d_out, int out_size, void* d_ws, size_t ws_size,
                              hipStream_t stream) {
  (void)in_sizes; (void)n_in; (void)out_size; (void)ws_size;
  const float* X   = (const float*)d_in[0];
  const float* W0  = (const float*)d_in[1];
  const float* BN0g=(const float*)d_in[2], *BN0b=(const float*)d_in[3], *BN0m=(const float*)d_in[4], *BN0v=(const float*)d_in[5];
  const float* W1  = (const float*)d_in[6];
  const float* BN1g=(const float*)d_in[7], *BN1b=(const float*)d_in[8], *BN1m=(const float*)d_in[9], *BN1v=(const float*)d_in[10];
  const float* W2  = (const float*)d_in[11];
  const float* BN2g=(const float*)d_in[12],*BN2b=(const float*)d_in[13],*BN2m=(const float*)d_in[14],*BN2v=(const float*)d_in[15];
  const float* W3  = (const float*)d_in[16];
  const float* BN3g=(const float*)d_in[17],*BN3b=(const float*)d_in[18],*BN3m=(const float*)d_in[19],*BN3v=(const float*)d_in[20];
  const float* W4  = (const float*)d_in[21];
  const float* BN4g=(const float*)d_in[22],*BN4b=(const float*)d_in[23],*BN4m=(const float*)d_in[24],*BN4v=(const float*)d_in[25];
  const float* W5  = (const float*)d_in[26];
  const float* BN5g=(const float*)d_in[27],*BN5b=(const float*)d_in[28],*BN5m=(const float*)d_in[29],*BN5v=(const float*)d_in[30];
  const float* FC1 = (const float*)d_in[31];
  const float* BF1g=(const float*)d_in[32],*BF1b=(const float*)d_in[33],*BF1m=(const float*)d_in[34],*BF1v=(const float*)d_in[35];
  const float* FC2 = (const float*)d_in[36];
  const float* BF2g=(const float*)d_in[37],*BF2b=(const float*)d_in[38],*BF2m=(const float*)d_in[39],*BF2v=(const float*)d_in[40];
  const float* FC3 = (const float*)d_in[41];
  const float* BF3m=(const float*)d_in[42],*BF3v=(const float*)d_in[43];
  float* OUT = (float*)d_out;

  char* wsb = (char*)d_ws; size_t off = 0;
  auto alloc = [&](size_t bytes) -> void* {
    void* p = wsb + off; off = (off + bytes + 255) & ~(size_t)255; return p;
  };
  int8_t* A0    = (int8_t*)alloc(128L*64*32*32);
  int8_t* X3A   = (int8_t*)alloc(128L*64*32*32);
  int8_t* A1    = (int8_t*)alloc(128L*64*16*16);
  int8_t* A2    = (int8_t*)alloc(128L*128*16*16);
  int8_t* A3    = (int8_t*)alloc(128L*128*8*8);
  int8_t* A4    = (int8_t*)alloc(128L*256*8*8);
  int8_t* A5    = (int8_t*)alloc(128L*256*4*4);
  int8_t* AFC1  = (int8_t*)alloc(128L*4160);
  int8_t* AFC1T = (int8_t*)alloc(128L*1408);
  int8_t* A6    = (int8_t*)alloc(128L*512);
  int8_t* A7    = (int8_t*)alloc(128L*512);
  int8_t* WB1   = (int8_t*)alloc(64L*576);
  int8_t* WB1T  = (int8_t*)alloc(64L*192);
  int8_t* WB2   = (int8_t*)alloc(128L*576);
  int8_t* WB2T  = (int8_t*)alloc(128L*192);
  int8_t* WB3   = (int8_t*)alloc(128L*1152);
  int8_t* WB4   = (int8_t*)alloc(256L*1152);
  int8_t* WB5   = (int8_t*)alloc(256L*2304);
  int8_t* WFC1  = (int8_t*)alloc(512L*4160);
  int8_t* WFC1T = (int8_t*)alloc(512L*1408);
  int8_t* WFC2  = (int8_t*)alloc(512L*512);
  int8_t* WFC3  = (int8_t*)alloc(1024L*512);
  int*    YF    = (int*)alloc(131072L*64*4);
  int*    YT    = (int*)alloc(131072L*64*4);

  auto nb = [](long n) { return (unsigned)((n + 255) / 256); };

  k_pack_sign<<<nb(64L*576),   256, 0, stream>>>(WB1,  W1,  64, 64, 576, 576);
  k_pack_trip_f<<<nb(64L*192), 256, 0, stream>>>(WB1T, W1,  64, 192, 192, 576);
  k_pack_sign<<<nb(128L*576),  256, 0, stream>>>(WB2,  W2, 128,128, 576, 576);
  k_pack_trip_f<<<nb(128L*192),256, 0, stream>>>(WB2T, W2, 128, 192, 192, 576);
  k_pack_sign<<<nb(128L*1152), 256, 0, stream>>>(WB3,  W3, 128,128,1152,1152);
  k_pack_sign<<<nb(256L*1152), 256, 0, stream>>>(WB4,  W4, 256,256,1152,1152);
  k_pack_sign<<<nb(256L*2304), 256, 0, stream>>>(WB5,  W5, 256,256,2304,2304);
  k_pack_sign<<<nb(512L*4160), 256, 0, stream>>>(WFC1, FC1,512,512,4160,4098);
  k_pack_trip_f<<<nb(512L*1408),256,0, stream>>>(WFC1T,FC1,512,1366,1408,4098);
  k_pack_sign<<<nb(512L*512),  256, 0, stream>>>(WFC2, FC2,512,512, 512, 512);
  k_pack_sign<<<nb(1024L*512), 256, 0, stream>>>(WFC3, FC3,1024,1000,512,512);

  k_conv0<<<nb(128L*64*32*32), 256, 0, stream>>>(A0, X, W0, BN0g, BN0b, BN0m, BN0v);

  k_x3<<<nb(128L*64*32*32), 256, 0, stream>>>(X3A, A0, 128*64, 32, 32);
  k_bgemm<3,3><<<dim3(131072/64, 1), 256, 0, stream>>>(YF, A0,  WB1, 128, 64,32,32, 1,1, 576, 576, 64);
  k_bgemm<3,1><<<dim3(131072/64, 1), 256, 0, stream>>>(YT, X3A, WB1T,128, 64,32,32, 1,0, 192, 192, 64);
  k_thresh_conv<<<nb(128L*64*16*16), 256, 0, stream>>>(A1, YF, YT, 128, 64, 64, 32,32, 2, 0.5f, BN1g,BN1b,BN1m,BN1v);

  k_x3<<<nb(128L*64*16*16), 256, 0, stream>>>(X3A, A1, 128*64, 16, 16);
  k_bgemm<3,3><<<dim3(32768/64, 2), 256, 0, stream>>>(YF, A1,  WB2, 128, 64,16,16, 1,1, 576, 576, 128);
  k_bgemm<3,1><<<dim3(32768/64, 2), 256, 0, stream>>>(YT, X3A, WB2T,128, 64,16,16, 1,0, 192, 192, 128);
  k_thresh_conv<<<nb(128L*128*16*16), 256, 0, stream>>>(A2, YF, YT, 128, 128, 128, 16,16, 1, 0.5f, BN2g,BN2b,BN2m,BN2v);

  k_bgemm<3,3><<<dim3(32768/64, 2), 256, 0, stream>>>(YF, A2, WB3, 128, 128,16,16, 1,1, 1152, 1152, 128);
  k_thresh_conv<<<nb(128L*128*8*8), 256, 0, stream>>>(A3, YF, (const int*)nullptr, 128, 128, 128, 16,16, 2, 1.0f, BN3g,BN3b,BN3m,BN3v);

  k_bgemm<3,3><<<dim3(8192/64, 4), 256, 0, stream>>>(YF, A3, WB4, 128, 128,8,8, 1,1, 1152, 1152, 256);
  k_thresh_conv<<<nb(128L*256*8*8), 256, 0, stream>>>(A4, YF, (const int*)nullptr, 128, 256, 256, 8,8, 1, 1.0f, BN4g,BN4b,BN4m,BN4v);

  k_bgemm<3,3><<<dim3(8192/64, 4), 256, 0, stream>>>(YF, A4, WB5, 128, 256,8,8, 1,1, 2304, 2304, 256);
  k_thresh_conv<<<nb(128L*256*4*4), 256, 0, stream>>>(A5, YF, (const int*)nullptr, 128, 256, 256, 8,8, 2, 1.0f, BN5g,BN5b,BN5m,BN5v);

  k_build_afc1<<<nb(128L*4160), 256, 0, stream>>>(AFC1, A5);
  k_trip_i8<<<nb(128L*1408), 256, 0, stream>>>(AFC1T, AFC1, 128, 1366, 1408, 4160);
  k_bgemm<1,1><<<dim3(128/64, 8), 256, 0, stream>>>(YF, AFC1,  WFC1, 128, 4160,1,1, 0,0, 4160, 4160, 512);
  k_bgemm<1,1><<<dim3(128/64, 8), 256, 0, stream>>>(YT, AFC1T, WFC1T,128, 1408,1,1, 0,0, 1408, 1408, 512);
  k_thresh_fc<<<nb(128L*512), 256, 0, stream>>>(A6, YF, YT, 128, 512, 512, 0.5f, BF1g,BF1b,BF1m,BF1v);

  k_bgemm<1,1><<<dim3(128/64, 8), 256, 0, stream>>>(YF, A6, WFC2, 128, 512,1,1, 0,0, 512, 512, 512);
  k_thresh_fc<<<nb(128L*512), 256, 0, stream>>>(A7, YF, (const int*)nullptr, 128, 512, 512, 1.0f, BF2g,BF2b,BF2m,BF2v);

  k_bgemm<1,1><<<dim3(128/64, 16), 256, 0, stream>>>(YF, A7, WFC3, 128, 512,1,1, 0,0, 512, 512, 1024);
  k_lsm<<<128, 256, 0, stream>>>(OUT, YF, BF3m, BF3v, 1000, 1024);
}
